// Mamba2_time_series_48455821034036
// MI455X (gfx1250) — compile-verified, matrix operand order checked by reading, not run on hardware
//
#include <hip/hip_runtime.h>
#include <hip/hip_bf16.h>

typedef __attribute__((ext_vector_type(16))) _Float16 v16h;
typedef __attribute__((ext_vector_type(8)))  float    v8f;
typedef __attribute__((ext_vector_type(8)))  unsigned int v8u;
typedef int vint4_t __attribute__((vector_size(16)));

#define B_ 8
#define T3 1024
#define M_ROWS (B_ * T3)        /* 8192 */
#define D_MODEL 768
#define D_INNER 1536
#define D_STATE 128
#define NHEADS 12
#define HEADDIM 128
#define CONV_DIM (D_INNER + 2 * D_STATE)          /* 1792 */
#define D_IN_PROJ (2*D_INNER + 2*D_STATE + NHEADS) /* 3340 */

#if __has_builtin(__builtin_amdgcn_global_load_async_to_lds_b128) && \
    __has_builtin(__builtin_amdgcn_s_wait_asynccnt)
#define HAVE_ASYNC_LDS 1
#else
#define HAVE_ASYNC_LDS 0
#endif

#define AS1V(p) ((__attribute__((address_space(1))) vint4_t*)(p))
#define AS3V(p) ((__attribute__((address_space(3))) vint4_t*)(p))

__device__ __forceinline__ float gelu_f(float x) {
    return 0.5f * x * (1.0f + erff(x * 0.70710678118654752f));
}
__device__ __forceinline__ float silu_f(float x) {
    return x / (1.0f + expf(-x));
}
__device__ __forceinline__ float bn_f(float x, float g, float b, float m, float v) {
    return (x - m) * rsqrtf(v + 1e-5f) * g + b;
}

/* ---------------- conv stem ---------------- */

__global__ __launch_bounds__(256) void conv1_k(
    const float* __restrict__ x, const float* __restrict__ w, const float* __restrict__ bias,
    const float* __restrict__ g, const float* __restrict__ bb,
    const float* __restrict__ mm, const float* __restrict__ vv,
    float* __restrict__ out)
{
    int bo = blockIdx.x; int b = bo / 192, o = bo % 192;
    int t = blockIdx.y * 256 + threadIdx.x;        /* T = 2048 */
    float s = bias[o];
    for (int i = 0; i < 12; ++i) {
        const float* xp = x + ((size_t)b * 12 + i) * 4096;
        const float* wp = w + (o * 12 + i) * 5;
        #pragma unroll
        for (int k = 0; k < 5; ++k) {
            int pos = 2 * t - 2 + k;
            if (pos >= 0 && pos < 4096) s += xp[pos] * wp[k];
        }
    }
    s = bn_f(gelu_f(s), g[o], bb[o], mm[o], vv[o]);
    out[((size_t)b * 192 + o) * 2048 + t] = s;
}

__global__ __launch_bounds__(256) void conv2_k(
    const float* __restrict__ in, const float* __restrict__ w, const float* __restrict__ bias,
    const float* __restrict__ g, const float* __restrict__ bb,
    const float* __restrict__ mm, const float* __restrict__ vv,
    float* __restrict__ out)
{
    int bo = blockIdx.x; int b = bo / 384, o = bo % 384;
    int t = blockIdx.y * 256 + threadIdx.x;        /* T = 1024 */
    float s = bias[o];
    for (int i = 0; i < 192; ++i) {
        const float* xp = in + ((size_t)b * 192 + i) * 2048;
        const float* wp = w + (o * 192 + i) * 3;
        #pragma unroll
        for (int k = 0; k < 3; ++k) {
            int pos = 2 * t - 1 + k;
            if (pos >= 0 && pos < 2048) s += xp[pos] * wp[k];
        }
    }
    s = bn_f(gelu_f(s), g[o], bb[o], mm[o], vv[o]);
    out[((size_t)b * 384 + o) * 1024 + t] = s;
}

__global__ __launch_bounds__(256) void conv3_k(
    const float* __restrict__ in, const float* __restrict__ w, const float* __restrict__ bias,
    const float* __restrict__ g, const float* __restrict__ bb,
    const float* __restrict__ mm, const float* __restrict__ vv,
    _Float16* __restrict__ u16)   /* (b, t, 768) f16, A matrix for in_proj */
{
    int bo = blockIdx.x; int b = bo / 768, o = bo % 768;
    int t = blockIdx.y * 256 + threadIdx.x;        /* T = 1024 */
    float s = bias[o];
    for (int i = 0; i < 384; ++i) {
        const float* xp = in + ((size_t)b * 384 + i) * 1024;
        const float* wp = w + (o * 384 + i) * 3;
        #pragma unroll
        for (int k = 0; k < 3; ++k) {
            int pos = t - 1 + k;
            if (pos >= 0 && pos < 1024) s += xp[pos] * wp[k];
        }
    }
    s = bn_f(gelu_f(s), g[o], bb[o], mm[o], vv[o]);
    u16[((size_t)b * 1024 + t) * 768 + o] = (_Float16)s;
}

/* ---------------- weight f32 -> f16 ---------------- */

__global__ __launch_bounds__(256) void cvt_w_k(
    const float* __restrict__ w1, _Float16* __restrict__ o1, int n1,
    const float* __restrict__ w2, _Float16* __restrict__ o2, int n2)
{
    int i = blockIdx.x * 256 + threadIdx.x;
    if (i < n1) o1[i] = (_Float16)w1[i];
    if (i < n2) o2[i] = (_Float16)w2[i];
}

/* ---------------- WMMA GEMM: C[M,N] = A[M,K] * W[N,K]^T ----------------
   block = 256 threads (8 waves), tile 32(M) x 128(N), K-tile 64.
   Waves arranged 2(M) x 4(N); per K-tile each wave: 2 A-frags, 4 B-frags, 4 WMMAs.
   Tiles staged via GLOBAL_LOAD_ASYNC_TO_LDS_B128 (ASYNCcnt) when available.
   LDS row stride = 36 dwords (144B): 16B-aligned for B128 writes and
   36*r mod 64 is distinct for r=0..15 -> conflict-free fragment gathers. */

#define LDSTRIDE 36

__global__ __launch_bounds__(256) void gemm_wmma_k(
    const _Float16* __restrict__ A, const _Float16* __restrict__ W,
    float* __restrict__ C, int M, int N, int K)
{
    __shared__ unsigned int As[32][LDSTRIDE];
    __shared__ unsigned int Ws[128][LDSTRIDE];

    const int tid  = threadIdx.x;
    const int lane = tid & 31, wave = tid >> 5;
    const int wm = wave >> 2, wn = wave & 3;
    const int half = lane >> 4, row = lane & 15;
    const int m0 = blockIdx.x * 32;
    const int n0 = blockIdx.y * 128;

    /* staging map: 8 B128 chunks per 64-f16 row */
    const int sr = tid >> 3;          /* 0..31  (A tile row)   */
    const int sc = tid & 7;           /* 0..7   (16B chunk)    */

    v8f acc0 = {}; v8f acc1 = {};

    for (int k0 = 0; k0 < K; k0 += 64) {
        /* ---- A tile: 32 rows x 64 f16, one B128 chunk per thread ---- */
        {
            const char* g = (const char*)A + ((size_t)(m0 + sr) * K + k0) * 2 + sc * 16;
            char* l = (char*)&As[sr][0] + sc * 16;
#if HAVE_ASYNC_LDS
            __builtin_amdgcn_global_load_async_to_lds_b128(AS1V(g), AS3V(l), 0, 0);
#else
            uint4 v = *(const uint4*)g;
            *(uint4*)l = v;
#endif
        }
        /* ---- W tile: 128 rows x 64 f16, four B128 chunks per thread ---- */
#if HAVE_ASYNC_LDS
        #pragma unroll
        for (int j = 0; j < 4; ++j) {
            int r = sr + j * 32;
            int n = n0 + r; if (n >= N) n = N - 1;   /* OOB cols never stored */
            const char* g = (const char*)W + ((size_t)n * K + k0) * 2 + sc * 16;
            char* l = (char*)&Ws[r][0] + sc * 16;
            __builtin_amdgcn_global_load_async_to_lds_b128(AS1V(g), AS3V(l), 0, 0);
        }
        __builtin_amdgcn_s_wait_asynccnt(0);
#else
        uint4 tmp[4];
        #pragma unroll
        for (int j = 0; j < 4; ++j) {
            int n = n0 + sr + j * 32; if (n >= N) n = N - 1;
            tmp[j] = *(const uint4*)((const char*)W + ((size_t)n * K + k0) * 2 + sc * 16);
        }
        #pragma unroll
        for (int j = 0; j < 4; ++j)
            *(uint4*)((char*)&Ws[sr + j * 32][0] + sc * 16) = tmp[j];
#endif
        __syncthreads();

        /* fragments per the 16-bit 16x32 lane layout:
           lane half selects K base 0/8; VGPR 0..3 => K+{0,2,4,6}, VGPR 4..7 => K+{16,18,20,22} */
        v8u au0, au1, b00, b01, b10, b11;
        #pragma unroll
        for (int i = 0; i < 8; ++i) {
            int c = (((half << 3) + ((i < 4) ? (i << 1) : (8 + (i << 1)))) >> 1);
            au0[i] = As[wm * 16 + row][c];
            au1[i] = As[wm * 16 + row][c + 16];
            b00[i] = Ws[wn * 32 + row][c];
            b01[i] = Ws[wn * 32 + 16 + row][c];
            b10[i] = Ws[wn * 32 + row][c + 16];
            b11[i] = Ws[wn * 32 + 16 + row][c + 16];
        }
        v16h a0 = __builtin_bit_cast(v16h, au0);
        v16h a1 = __builtin_bit_cast(v16h, au1);
        acc0 = __builtin_amdgcn_wmma_f32_16x16x32_f16(false, a0, false,
                 __builtin_bit_cast(v16h, b00), (short)0, acc0, false, false);
        acc1 = __builtin_amdgcn_wmma_f32_16x16x32_f16(false, a0, false,
                 __builtin_bit_cast(v16h, b01), (short)0, acc1, false, false);
        acc0 = __builtin_amdgcn_wmma_f32_16x16x32_f16(false, a1, false,
                 __builtin_bit_cast(v16h, b10), (short)0, acc0, false, false);
        acc1 = __builtin_amdgcn_wmma_f32_16x16x32_f16(false, a1, false,
                 __builtin_bit_cast(v16h, b11), (short)0, acc1, false, false);
        __syncthreads();
    }

    /* C/D layout: vgpr r, lane L -> M = r + 8*(L>=16), N = L&15 */
    int mBase = m0 + wm * 16 + half * 8;
    int nA = n0 + wn * 32 + row;
    int nB = nA + 16;
    #pragma unroll
    for (int r = 0; r < 8; ++r) {
        if (nA < N) C[(size_t)(mBase + r) * N + nA] = acc0[r];
        if (nB < N) C[(size_t)(mBase + r) * N + nB] = acc1[r];
    }
}

/* ---------------- dt softplus + dA ---------------- */

__global__ __launch_bounds__(256) void dtda_k(
    const float* __restrict__ zx, const float* __restrict__ dt_bias,
    const float* __restrict__ A_log, float* __restrict__ dt, float* __restrict__ dA)
{
    int i = blockIdx.x * 256 + threadIdx.x;   /* 8192*12 */
    if (i >= M_ROWS * NHEADS) return;
    int row = i / NHEADS, h = i % NHEADS;
    float x = zx[(size_t)row * D_IN_PROJ + (D_INNER + CONV_DIM) + h] + dt_bias[h];
    float sp = (x > 20.f) ? x : log1pf(expf(x));
    dt[i] = sp;
    dA[i] = expf(sp * (-expf(A_log[h])));
}

/* ---------------- depthwise causal conv + SiLU ---------------- */

__global__ __launch_bounds__(256) void dwconv_k(
    const float* __restrict__ zx, const float* __restrict__ cw,
    const float* __restrict__ cb, float* __restrict__ xout)
{
    int i = blockIdx.x * 256 + threadIdx.x;   /* 8192*1792 */
    if (i >= M_ROWS * CONV_DIM) return;
    int c = i % CONV_DIM; int row = i / CONV_DIM;
    int t = row % T3; int b = row / T3;
    float s = cb[c];
    #pragma unroll
    for (int k = 0; k < 4; ++k) {
        int tt = t - 3 + k;
        if (tt >= 0)
            s += zx[((size_t)(b * T3 + tt)) * D_IN_PROJ + D_INNER + c] * cw[c * 4 + k];
    }
    xout[(size_t)row * CONV_DIM + c] = silu_f(s);
}

/* ---------------- SSM sequential scan ----------------
   one block per (b,h), 128 lanes = head dim; 128-float state lives in VGPRs */

__global__ __launch_bounds__(128) void scan_k(
    const float* __restrict__ xBC, const float* __restrict__ dt,
    const float* __restrict__ dA, const float* __restrict__ Dp,
    float* __restrict__ y)
{
    int b = blockIdx.x, h = blockIdx.y;
    int p = threadIdx.x;
    __shared__ float Bs[D_STATE];
    __shared__ float Cs[D_STATE];

    float st[D_STATE];
    #pragma unroll
    for (int n = 0; n < D_STATE; ++n) st[n] = 0.f;

    const float Dh = Dp[h];
    for (int t = 0; t < T3; ++t) {
        const float* rw = xBC + ((size_t)(b * T3 + t)) * CONV_DIM;
        float xv = rw[h * HEADDIM + p];
        Bs[p] = rw[D_INNER + p];
        Cs[p] = rw[D_INNER + D_STATE + p];
        __syncthreads();
        int sidx = (b * T3 + t) * NHEADS + h;
        float dAv = dA[sidx];
        float dx  = dt[sidx] * xv;
        float acc = 0.f;
        #pragma unroll
        for (int n = 0; n < D_STATE; ++n) {
            st[n] = st[n] * dAv + dx * Bs[n];
            acc += st[n] * Cs[n];
        }
        y[((size_t)(b * T3 + t)) * D_INNER + h * HEADDIM + p] = acc + Dh * xv;
        __syncthreads();
    }
}

/* ---------------- gate (silu(z)) + RMSNorm -> f16 ---------------- */

__global__ __launch_bounds__(256) void gatenorm_k(
    const float* __restrict__ y, const float* __restrict__ zx,
    const float* __restrict__ nw, _Float16* __restrict__ y16)
{
    __shared__ float gb[D_INNER];
    __shared__ float red[256];
    int row = blockIdx.x; int tid = threadIdx.x;
    float acc = 0.f;
    for (int c = tid; c < D_INNER; c += 256) {
        float yv = y[(size_t)row * D_INNER + c];
        float zv = zx[(size_t)row * D_IN_PROJ + c];
        float gv = yv * silu_f(zv);
        gb[c] = gv;
        acc += gv * gv;
    }
    red[tid] = acc; __syncthreads();
    for (int s = 128; s > 0; s >>= 1) {
        if (tid < s) red[tid] += red[tid + s];
        __syncthreads();
    }
    float r = rsqrtf(red[0] * (1.0f / D_INNER) + 1e-5f);
    for (int c = tid; c < D_INNER; c += 256)
        y16[(size_t)row * D_INNER + c] = (_Float16)(gb[c] * r * nw[c]);
}

/* ---------------- mean over time ---------------- */

__global__ __launch_bounds__(256) void mean_k(const float* __restrict__ op, float* __restrict__ rep)
{
    int bc = blockIdx.x; int b = bc / D_MODEL, c = bc % D_MODEL;
    __shared__ float red[256];
    float acc = 0.f;
    for (int t = threadIdx.x; t < T3; t += 256)
        acc += op[((size_t)(b * T3 + t)) * D_MODEL + c];
    red[threadIdx.x] = acc; __syncthreads();
    for (int s = 128; s > 0; s >>= 1) {
        if (threadIdx.x < s) red[threadIdx.x] += red[threadIdx.x + s];
        __syncthreads();
    }
    if (threadIdx.x == 0) rep[bc] = red[0] * (1.0f / T3);
}

/* ---------------- classifier head ---------------- */

__global__ __launch_bounds__(256) void fc1_k(
    const float* __restrict__ rep, const float* __restrict__ w, const float* __restrict__ bias,
    const float* __restrict__ g, const float* __restrict__ bb,
    const float* __restrict__ mm, const float* __restrict__ vv,
    float* __restrict__ h1)
{
    int i = blockIdx.x * 256 + threadIdx.x;   /* 8*1024 */
    if (i >= B_ * 1024) return;
    int b = i / 1024, j = i % 1024;
    const float* r = rep + b * D_MODEL;
    const float* wp = w + (size_t)j * D_MODEL;
    float s = bias[j];
    for (int c = 0; c < D_MODEL; ++c) s += r[c] * wp[c];
    s = bn_f(s, g[j], bb[j], mm[j], vv[j]);
    h1[i] = fmaxf(s, 0.f);
}

__global__ __launch_bounds__(64) void fc2_k(
    const float* __restrict__ h1, const float* __restrict__ w,
    const float* __restrict__ bias, float* __restrict__ out)
{
    int i = blockIdx.x * 64 + threadIdx.x;    /* 8*27 */
    if (i >= B_ * 27) return;
    int b = i / 27, o = i % 27;
    float s = bias[o];
    const float* hp = h1 + b * 1024;
    const float* wp = w + (size_t)o * 1024;
    for (int j = 0; j < 1024; ++j) s += hp[j] * wp[j];
    out[i] = s;
}

/* ---------------- launch ---------------- */

extern "C" void kernel_launch(void* const* d_in, const int* in_sizes, int n_in,
                              void* d_out, int out_size, void* d_ws, size_t ws_size,
                              hipStream_t stream)
{
    const float* x       = (const float*)d_in[0];
    const float* w1      = (const float*)d_in[1];
    const float* b1      = (const float*)d_in[2];
    const float* bn1_g   = (const float*)d_in[3];
    const float* bn1_b   = (const float*)d_in[4];
    const float* bn1_m   = (const float*)d_in[5];
    const float* bn1_v   = (const float*)d_in[6];
    const float* w2      = (const float*)d_in[7];
    const float* b2      = (const float*)d_in[8];
    const float* bn2_g   = (const float*)d_in[9];
    const float* bn2_b   = (const float*)d_in[10];
    const float* bn2_m   = (const float*)d_in[11];
    const float* bn2_v   = (const float*)d_in[12];
    const float* w3      = (const float*)d_in[13];
    const float* b3      = (const float*)d_in[14];
    const float* bn3_g   = (const float*)d_in[15];
    const float* bn3_b   = (const float*)d_in[16];
    const float* bn3_m   = (const float*)d_in[17];
    const float* bn3_v   = (const float*)d_in[18];
    const float* in_proj_w  = (const float*)d_in[19];
    const float* conv_w  = (const float*)d_in[20];
    const float* conv_b  = (const float*)d_in[21];
    const float* dt_bias = (const float*)d_in[22];
    const float* A_log   = (const float*)d_in[23];
    const float* Dp      = (const float*)d_in[24];
    const float* norm_w  = (const float*)d_in[25];
    const float* out_proj_w = (const float*)d_in[26];
    const float* fc1_w   = (const float*)d_in[27];
    const float* fc1_b   = (const float*)d_in[28];
    const float* bnc_g   = (const float*)d_in[29];
    const float* bnc_b   = (const float*)d_in[30];
    const float* bnc_m   = (const float*)d_in[31];
    const float* bnc_v   = (const float*)d_in[32];
    const float* fc2_w   = (const float*)d_in[33];
    const float* fc2_b   = (const float*)d_in[34];
    float* out = (float*)d_out;

    /* workspace carve-up */
    char* wp8 = (char*)d_ws;
    auto carve = [&](size_t bytes) -> void* {
        void* p = (void*)wp8;
        wp8 += (bytes + 255) & ~(size_t)255;
        return p;
    };
    _Float16* u16   = (_Float16*)carve((size_t)M_ROWS * D_MODEL * 2);
    _Float16* w16a  = (_Float16*)carve((size_t)D_IN_PROJ * D_MODEL * 2);
    _Float16* w16b  = (_Float16*)carve((size_t)D_MODEL * D_INNER * 2);
    float* h1c      = (float*)carve((size_t)B_ * 192 * 2048 * 4);
    float* h2c      = (float*)carve((size_t)B_ * 384 * 1024 * 4);
    float* zxbcdt   = (float*)carve((size_t)M_ROWS * D_IN_PROJ * 4);
    float* dtb      = (float*)carve((size_t)M_ROWS * NHEADS * 4);
    float* dAb      = (float*)carve((size_t)M_ROWS * NHEADS * 4);
    float* xBCa     = (float*)carve((size_t)M_ROWS * CONV_DIM * 4);
    float* ybuf     = (float*)carve((size_t)M_ROWS * D_INNER * 4);
    _Float16* y16   = (_Float16*)carve((size_t)M_ROWS * D_INNER * 2);
    float* oproj    = (float*)carve((size_t)M_ROWS * D_MODEL * 4);
    float* rep      = (float*)carve((size_t)B_ * D_MODEL * 4);
    float* h1fc     = (float*)carve((size_t)B_ * 1024 * 4);

    /* weights to f16 (in_proj_w: 3340x768, out_proj_w: 768x1536) */
    {
        int n1 = D_IN_PROJ * D_MODEL;
        int n2 = D_MODEL * D_INNER;
        int nmax = n1 > n2 ? n1 : n2;
        cvt_w_k<<<(nmax + 255) / 256, 256, 0, stream>>>(in_proj_w, w16a, n1, out_proj_w, w16b, n2);
    }

    /* conv stem */
    conv1_k<<<dim3(B_ * 192, 2048 / 256), 256, 0, stream>>>(x, w1, b1, bn1_g, bn1_b, bn1_m, bn1_v, h1c);
    conv2_k<<<dim3(B_ * 384, 1024 / 256), 256, 0, stream>>>(h1c, w2, b2, bn2_g, bn2_b, bn2_m, bn2_v, h2c);
    conv3_k<<<dim3(B_ * 768, 1024 / 256), 256, 0, stream>>>(h2c, w3, b3, bn3_g, bn3_b, bn3_m, bn3_v, u16);

    /* in-projection GEMM: (8192 x 768) @ (3340 x 768)^T */
    {
        dim3 grid(M_ROWS / 32, (D_IN_PROJ + 127) / 128);
        gemm_wmma_k<<<grid, 256, 0, stream>>>(u16, w16a, zxbcdt, M_ROWS, D_IN_PROJ, D_MODEL);
    }

    /* dt softplus + dA */
    dtda_k<<<(M_ROWS * NHEADS + 255) / 256, 256, 0, stream>>>(zxbcdt, dt_bias, A_log, dtb, dAb);

    /* depthwise causal conv + SiLU */
    dwconv_k<<<(M_ROWS * CONV_DIM + 255) / 256, 256, 0, stream>>>(zxbcdt, conv_w, conv_b, xBCa);

    /* SSM scan */
    scan_k<<<dim3(B_, NHEADS), 128, 0, stream>>>(xBCa, dtb, dAb, Dp, ybuf);

    /* gate + RMSNorm -> f16 */
    gatenorm_k<<<M_ROWS, 256, 0, stream>>>(ybuf, zxbcdt, norm_w, y16);

    /* out-projection GEMM: (8192 x 1536) @ (768 x 1536)^T */
    {
        dim3 grid(M_ROWS / 32, D_MODEL / 128);
        gemm_wmma_k<<<grid, 256, 0, stream>>>(y16, w16b, oproj, M_ROWS, D_MODEL, D_INNER);
    }

    /* mean over time */
    mean_k<<<B_ * D_MODEL, 256, 0, stream>>>(oproj, rep);

    /* classifier head */
    fc1_k<<<(B_ * 1024 + 255) / 256, 256, 0, stream>>>(rep, fc1_w, fc1_b, bnc_g, bnc_b, bnc_m, bnc_v, h1fc);
    fc2_k<<<(B_ * 27 + 63) / 64, 64, 0, stream>>>(h1fc, fc2_w, fc2_b, out);
    (void)in_sizes; (void)n_in; (void)out_size; (void)ws_size;
}
